// TransitionGNN_25718264168600
// MI455X (gfx1250) — matrix operand order repaired and checked by reading, not compiled, not run on hardware
//
#include <hip/hip_runtime.h>
#include <hip/hip_bf16.h>

typedef __attribute__((ext_vector_type(16))) _Float16 v16h;
typedef __attribute__((ext_vector_type(4)))  _Float16 v4h;
typedef __attribute__((ext_vector_type(8)))  float    v8f;
typedef __attribute__((ext_vector_type(4)))  int      v4i;

typedef __attribute__((address_space(1))) v4i v4i_global;
typedef __attribute__((address_space(3))) v4i v4i_lds;

#define E_EDGES 122880   // 512 * 240
#define BN_NODES 8192    // 512 * 16

#if __has_builtin(__builtin_amdgcn_global_load_async_to_lds_b128)
#define HAVE_ASYNC_LDS 1
#else
#define HAVE_ASYNC_LDS 0
#endif

__device__ __forceinline__ void wait_async_zero() {
#if __has_builtin(__builtin_amdgcn_s_wait_asynccnt)
    __builtin_amdgcn_s_wait_asynccnt(0);
#elif HAVE_ASYNC_LDS
    asm volatile("s_wait_asynccnt 0x0" ::: "memory");
#endif
}

__device__ __forceinline__ void async_cp_b128(const _Float16* g, _Float16* l) {
#if HAVE_ASYNC_LDS
    __builtin_amdgcn_global_load_async_to_lds_b128(
        (v4i_global*)(void*)g, (v4i_lds*)(void*)l, 0, 0);
#else
    *(uint4*)l = *(const uint4*)g;   // sync fallback (VGPR round-trip)
#endif
}

// ---------------------------------------------------------------------------
// Weight prep: cast fp32 [K,N] -> f16 transposed [N,Kp] (zero-pad K..Kp)
// ---------------------------------------------------------------------------
__global__ __launch_bounds__(256) void cast_transpose(const float* __restrict__ W,
                                                      _Float16* __restrict__ Wt,
                                                      int K, int N, int Kp) {
    int idx = blockIdx.x * 256 + threadIdx.x;
    if (idx >= N * Kp) return;
    int n = idx / Kp, k = idx % Kp;
    Wt[(size_t)n * Kp + k] = (k < K) ? (_Float16)W[(size_t)k * N + n] : (_Float16)0.0f;
}

// ---------------------------------------------------------------------------
// Edge feature build: Ef[e] = concat(states[row(e)], states[col(e)]) as f16
// edge list is np.nonzero(1-eye(16)) row-major => sorted by src
// ---------------------------------------------------------------------------
__global__ __launch_bounds__(256) void build_edges(const float* __restrict__ states,
                                                   _Float16* __restrict__ Ef) {
    int idx = blockIdx.x * 256 + threadIdx.x;      // one float4 per thread
    int e  = idx >> 5;                             // 32 float4 per edge row (128 f16)
    int c4 = (idx & 31) * 4;
    int b = e / 240, r = e % 240;
    int src = r / 15, j = r % 15;
    int dst = j + (j >= src ? 1 : 0);
    int node, c;
    if (c4 < 64) { node = b * 16 + src; c = c4; }
    else         { node = b * 16 + dst; c = c4 - 64; }
    float4 v = *(const float4*)(states + (size_t)node * 64 + c);
    v4h h; h.x = (_Float16)v.x; h.y = (_Float16)v.y; h.z = (_Float16)v.z; h.w = (_Float16)v.w;
    *(v4h*)(Ef + (size_t)e * 128 + c4) = h;
}

// ---------------------------------------------------------------------------
// Segment sum: agg[node][:] = sum_{j<15} edge_attr[node*15+j][:]
// ---------------------------------------------------------------------------
__global__ __launch_bounds__(256) void agg_reduce(const float* __restrict__ EA,
                                                  float* __restrict__ agg) {
    int idx = blockIdx.x * 256 + threadIdx.x;      // BN_NODES * 128 float4 lanes
    int node = idx >> 7;
    int c = (idx & 127) * 4;
    float4 s = {0.f, 0.f, 0.f, 0.f};
    const float* base = EA + (size_t)node * 15 * 512 + c;
#pragma unroll
    for (int j = 0; j < 15; ++j) {
        float4 v = *(const float4*)(base + (size_t)j * 512);
        s.x += v.x; s.y += v.y; s.z += v.z; s.w += v.w;
    }
    *(float4*)(agg + (size_t)node * 512 + c) = s;
}

// ---------------------------------------------------------------------------
// Node feature build: x = [states(64) | onehot-act(4) | agg(512) | pad(28)] f16
// ---------------------------------------------------------------------------
__global__ __launch_bounds__(256) void build_node(const float* __restrict__ states,
                                                  const int* __restrict__ action,
                                                  const float* __restrict__ agg,
                                                  _Float16* __restrict__ X) {
    int idx = blockIdx.x * 256 + threadIdx.x;
    if (idx >= BN_NODES * 608) return;
    int node = idx / 608, c = idx % 608;
    float v;
    if (c < 64)       v = states[(size_t)node * 64 + c];
    else if (c < 68)  { int b = node / 16, n = node % 16;
                        v = (action[b] == n * 4 + (c - 64)) ? 1.0f : 0.0f; }
    else if (c < 580) v = agg[(size_t)node * 512 + (c - 68)];
    else              v = 0.0f;
    X[(size_t)node * 608 + c] = (_Float16)v;
}

// ---------------------------------------------------------------------------
// LayerNorm(512) + ReLU, f32 in -> f16 out. One wave per row (wave32).
// ---------------------------------------------------------------------------
__global__ __launch_bounds__(256) void ln_relu(const float* __restrict__ X,
                                               const float* __restrict__ g,
                                               const float* __restrict__ bb,
                                               _Float16* __restrict__ Y, int rows) {
    int wave = threadIdx.x >> 5, lane = threadIdx.x & 31;
    int row = blockIdx.x * 8 + wave;
    if (row >= rows) return;
    const float* x = X + (size_t)row * 512;
    float4 v[4];
    float s = 0.f, ss = 0.f;
#pragma unroll
    for (int j = 0; j < 4; ++j) {
        v[j] = *(const float4*)(x + (j * 32 + lane) * 4);
        s  += v[j].x + v[j].y + v[j].z + v[j].w;
        ss += v[j].x*v[j].x + v[j].y*v[j].y + v[j].z*v[j].z + v[j].w*v[j].w;
    }
#pragma unroll
    for (int off = 16; off > 0; off >>= 1) {
        s  += __shfl_xor(s,  off, 32);
        ss += __shfl_xor(ss, off, 32);
    }
    float mean = s * (1.0f / 512.0f);
    float var  = ss * (1.0f / 512.0f) - mean * mean;
    float rstd = rsqrtf(var + 1e-5f);
#pragma unroll
    for (int j = 0; j < 4; ++j) {
        int c = (j * 32 + lane) * 4;
        float o0 = fmaxf((v[j].x - mean) * rstd * g[c+0] + bb[c+0], 0.f);
        float o1 = fmaxf((v[j].y - mean) * rstd * g[c+1] + bb[c+1], 0.f);
        float o2 = fmaxf((v[j].z - mean) * rstd * g[c+2] + bb[c+2], 0.f);
        float o3 = fmaxf((v[j].w - mean) * rstd * g[c+3] + bb[c+3], 0.f);
        v4h h; h.x = (_Float16)o0; h.y = (_Float16)o1; h.z = (_Float16)o2; h.w = (_Float16)o3;
        *(v4h*)(Y + (size_t)row * 512 + c) = h;
    }
}

// ---------------------------------------------------------------------------
// WMMA GEMM: C[M,N] = A[M,K](f16 rowmajor) * Bt[N,K](f16, pre-transposed) + bias
// EPI 0: relu -> f16 out.  EPI 1: f32 out (no relu).
// 256 threads = 8 waves (WAVES_M x WAVES_N); wave tile (WM*16) x (WN*16) via
// v_wmma_f32_16x16x32_f16. K stepped by 32 through DOUBLE-BUFFERED LDS filled
// with global_load_async_to_lds_b128 (ASYNCcnt), so the stage for step t+1
// overlaps the WMMAs of step t.
// ---------------------------------------------------------------------------
template<int BM, int BN, int BK, int WAVES_M, int WAVES_N, int WM, int WN, int EPI>
__global__ __launch_bounds__(256) void gemm_wmma(const _Float16* __restrict__ A,
                                                 const _Float16* __restrict__ Bt,
                                                 const float* __restrict__ bias,
                                                 void* __restrict__ out,
                                                 int M, int N, int K) {
    constexpr int LDA = BK + 8;                 // stagger LDS banks
    __shared__ _Float16 As[2][BM * LDA];
    __shared__ _Float16 Bs[2][BN * LDA];

    const int tid  = threadIdx.x;
    const int wave = tid >> 5, lane = tid & 31;
    const int wm = wave / WAVES_N, wn = wave % WAVES_N;
    const int wM0 = wm * WM * 16, wN0 = wn * WN * 16;
    const int bM0 = blockIdx.x * BM, bN0 = blockIdx.y * BN;

    v8f acc[WM][WN];
#pragma unroll
    for (int i = 0; i < WM; ++i)
#pragma unroll
        for (int j = 0; j < WN; ++j) { v8f z = {}; acc[i][j] = z; }

    const int l15 = lane & 15;
    const int hi  = lane >> 4;                  // 0: lanes 0-15, 1: lanes 16-31

    // issue all async b128 copies for the tile at column k0 into buffer `buf`
    auto stage = [&](int buf, int k0) {
        constexpr int AV = BM * BK / 8;
#pragma unroll 2
        for (int i = tid; i < AV; i += 256) {
            int row = i / (BK / 8), kc = (i % (BK / 8)) * 8;
            async_cp_b128(A + (size_t)(bM0 + row) * K + k0 + kc,
                          &As[buf][row * LDA + kc]);
        }
        constexpr int BV = BN * BK / 8;
#pragma unroll 2
        for (int i = tid; i < BV; i += 256) {
            int row = i / (BK / 8), kc = (i % (BK / 8)) * 8;
            async_cp_b128(Bt + (size_t)(bN0 + row) * K + k0 + kc,
                          &Bs[buf][row * LDA + kc]);
        }
    };

    stage(0, 0);
    int buf = 0;
    for (int k0 = 0; k0 < K; k0 += BK) {
        wait_async_zero();          // tile for this step landed in LDS
        __syncthreads();            // ...for every wave in the block
        if (k0 + BK < K)            // prefetch next step into the other buffer
            stage(buf ^ 1, k0 + BK);

        // A frags: lane l<16 -> row l, K {0-7,16-23}; lane l+16 -> K {8-15,24-31}
        v16h af[WM], bf[WN];
#pragma unroll
        for (int mt = 0; mt < WM; ++mt) {
            const _Float16* ap = &As[buf][(wM0 + mt * 16 + l15) * LDA + hi * 8];
#pragma unroll
            for (int i = 0; i < 8; ++i) { af[mt][i] = ap[i]; af[mt][8 + i] = ap[16 + i]; }
        }
        // B frags: lane l<16 -> col l, K {0-7,16-23}; lane l+16 -> col l-16, K {8-15,24-31}
#pragma unroll
        for (int nt = 0; nt < WN; ++nt) {
            const _Float16* bp = &Bs[buf][(wN0 + nt * 16 + l15) * LDA + hi * 8];
#pragma unroll
            for (int i = 0; i < 8; ++i) { bf[nt][i] = bp[i]; bf[nt][8 + i] = bp[16 + i]; }
        }
#pragma unroll
        for (int mt = 0; mt < WM; ++mt)
#pragma unroll
            for (int nt = 0; nt < WN; ++nt)
                acc[mt][nt] = __builtin_amdgcn_wmma_f32_16x16x32_f16(
                    false, af[mt], false, bf[nt], (short)0, acc[mt][nt], false, false);
        buf ^= 1;
    }

    // C layout: VGPR v -> row v (lanes 0-15) / v+8 (lanes 16-31); col = lane&15
#pragma unroll
    for (int nt = 0; nt < WN; ++nt) {
        int gcol = bN0 + wN0 + nt * 16 + l15;
        float bv = bias[gcol];
#pragma unroll
        for (int mt = 0; mt < WM; ++mt) {
#pragma unroll
            for (int v = 0; v < 8; ++v) {
                int grow = bM0 + wM0 + mt * 16 + v + hi * 8;
                float val = acc[mt][nt][v] + bv;
                if (EPI == 0) {
                    val = fmaxf(val, 0.0f);
                    ((_Float16*)out)[(size_t)grow * N + gcol] = (_Float16)val;
                } else {
                    ((float*)out)[(size_t)grow * N + gcol] = val;
                }
            }
        }
    }
}

// ---------------------------------------------------------------------------
extern "C" void kernel_launch(void* const* d_in, const int* in_sizes, int n_in,
                              void* d_out, int out_size, void* d_ws, size_t ws_size,
                              hipStream_t stream) {
    const float* states = (const float*)d_in[0];
    const int*   action = (const int*)  d_in[1];
    const float* ew1 = (const float*)d_in[2];  const float* eb1 = (const float*)d_in[3];
    const float* ew2 = (const float*)d_in[4];  const float* eb2 = (const float*)d_in[5];
    const float* eg  = (const float*)d_in[6];  const float* ebn = (const float*)d_in[7];
    const float* ew3 = (const float*)d_in[8];  const float* eb3 = (const float*)d_in[9];
    const float* nw1 = (const float*)d_in[10]; const float* nb1 = (const float*)d_in[11];
    const float* nw2 = (const float*)d_in[12]; const float* nb2 = (const float*)d_in[13];
    const float* ng  = (const float*)d_in[14]; const float* nbn = (const float*)d_in[15];
    const float* nw3 = (const float*)d_in[16]; const float* nb3 = (const float*)d_in[17];

    char* ws = (char*)d_ws;
    size_t off = 0;
    auto alloc = [&](size_t bytes) -> void* {
        void* p = ws + off;
        off = (off + bytes + 255) & ~(size_t)255;
        return p;
    };

    const size_t E = E_EDGES, BNn = BN_NODES;
    _Float16* ew1t = (_Float16*)alloc(512 * 128 * 2);
    _Float16* ew2t = (_Float16*)alloc(512 * 512 * 2);
    _Float16* ew3t = (_Float16*)alloc(512 * 512 * 2);
    _Float16* nw1t = (_Float16*)alloc(512 * 608 * 2);
    _Float16* nw2t = (_Float16*)alloc(512 * 512 * 2);
    _Float16* nw3t = (_Float16*)alloc(64 * 512 * 2);
    _Float16* Ef   = (_Float16*)alloc(E * 128 * 2);
    _Float16* h1   = (_Float16*)alloc(E * 512 * 2);   // reused for h2
    float*    t2   = (float*)   alloc(E * 512 * 4);   // pre-LN f32, reused for edge_attr
    float*    agg  = (float*)   alloc(BNn * 512 * 4);
    _Float16* xn   = (_Float16*)alloc(BNn * 608 * 2);
    _Float16* n1   = (_Float16*)alloc(BNn * 512 * 2); // reused for n2
    float*    tn   = (float*)   alloc(BNn * 512 * 4);

    // --- weight cast+transpose (f32 [K,N] -> f16 [N,Kp]) ---
    cast_transpose<<<(512 * 128 + 255) / 256, 256, 0, stream>>>(ew1, ew1t, 128, 512, 128);
    cast_transpose<<<(512 * 512 + 255) / 256, 256, 0, stream>>>(ew2, ew2t, 512, 512, 512);
    cast_transpose<<<(512 * 512 + 255) / 256, 256, 0, stream>>>(ew3, ew3t, 512, 512, 512);
    cast_transpose<<<(512 * 608 + 255) / 256, 256, 0, stream>>>(nw1, nw1t, 580, 512, 608);
    cast_transpose<<<(512 * 512 + 255) / 256, 256, 0, stream>>>(nw2, nw2t, 512, 512, 512);
    cast_transpose<<<(64 * 512 + 255) / 256, 256, 0, stream>>>(nw3, nw3t, 512, 64, 512);

    // --- edge path ---
    build_edges<<<(E * 32) / 256, 256, 0, stream>>>(states, Ef);
    gemm_wmma<128, 256, 32, 2, 4, 4, 4, 0>
        <<<dim3(E / 128, 2), 256, 0, stream>>>(Ef, ew1t, eb1, h1, (int)E, 512, 128);
    gemm_wmma<128, 256, 32, 2, 4, 4, 4, 1>
        <<<dim3(E / 128, 2), 256, 0, stream>>>(h1, ew2t, eb2, t2, (int)E, 512, 512);
    ln_relu<<<E / 8, 256, 0, stream>>>(t2, eg, ebn, h1, (int)E);           // h2 -> h1 region
    gemm_wmma<128, 256, 32, 2, 4, 4, 4, 1>
        <<<dim3(E / 128, 2), 256, 0, stream>>>(h1, ew3t, eb3, t2, (int)E, 512, 512); // edge_attr
    agg_reduce<<<(BNn * 128) / 256, 256, 0, stream>>>(t2, agg);

    // --- node path ---
    build_node<<<((unsigned)(BNn * 608) + 255) / 256, 256, 0, stream>>>(states, action, agg, xn);
    gemm_wmma<128, 256, 32, 2, 4, 4, 4, 0>
        <<<dim3(BNn / 128, 2), 256, 0, stream>>>(xn, nw1t, nb1, n1, (int)BNn, 512, 608);
    gemm_wmma<128, 256, 32, 2, 4, 4, 4, 1>
        <<<dim3(BNn / 128, 2), 256, 0, stream>>>(n1, nw2t, nb2, tn, (int)BNn, 512, 512);
    ln_relu<<<BNn / 8, 256, 0, stream>>>(tn, ng, nbn, n1, (int)BNn);       // n2 -> n1 region
    gemm_wmma<128, 64, 32, 4, 2, 2, 2, 1>
        <<<dim3(BNn / 128, 1), 256, 0, stream>>>(n1, nw3t, nb3, (float*)d_out, (int)BNn, 64, 512);
}
